// MotionOptimalTransportLoss_23407571764203
// MI455X (gfx1250) — matrix operand order repaired and checked by reading, not compiled, not run on hardware
//
#include <hip/hip_runtime.h>

// ---------------------------------------------------------------------------
// Types for CDNA5 WMMA (wave32, 16x16x32 bf16 -> f32)
// ---------------------------------------------------------------------------
typedef __attribute__((ext_vector_type(16))) __bf16 v16bf;
typedef __attribute__((ext_vector_type(8)))  __bf16 v8bf;
typedef __attribute__((ext_vector_type(8)))  float  v8f;

#define NP 1024          // padded point count (all levels)
#define BATCH 8
#define EPSV 1e-10f
#define BIGF 3.0e38f

#define WMMA_BF16(A, B, C) \
    __builtin_amdgcn_wmma_f32_16x16x32_bf16(false, (A), false, (B), (short)0, (C), false, false)

// ---------------------------------------------------------------------------
// Helpers
// ---------------------------------------------------------------------------
__device__ __forceinline__ unsigned short f2bf(float x) {
    unsigned u = __float_as_uint(x);
    unsigned r = u + 0x7fffu + ((u >> 16) & 1u);   // round-to-nearest-even
    return (unsigned short)(r >> 16);
}
__device__ __forceinline__ float bf2f(unsigned short h) {
    return __uint_as_float(((unsigned)h) << 16);
}
// order-preserving float -> uint key (for atomicMin on floats of any sign)
__device__ __forceinline__ unsigned f2key(float f) {
    unsigned u = __float_as_uint(f);
    return (u & 0x80000000u) ? ~u : (u | 0x80000000u);
}
__device__ __forceinline__ float key2f(unsigned k) {
    unsigned u = (k & 0x80000000u) ? (k & 0x7fffffffu) : ~k;
    return __uint_as_float(u);
}

// A fragment (16x32, M x K): lane L<16 owns row L with K {k0..k0+15};
// lane L>=16 owns row L-16 with K {k0+16..k0+31} (matches B fragment order).
__device__ __forceinline__ v16bf load_a_frag(const __bf16* base, int row0,
                                             int stride, int k0, int lane) {
    int m = lane & 15, half = lane >> 4;
    const __bf16* rp = base + (size_t)(row0 + m) * stride + k0 + 16 * half;
    v8bf lo = *reinterpret_cast<const v8bf*>(rp);
    v8bf hi = *reinterpret_cast<const v8bf*>(rp + 8);
    return __builtin_shufflevector(lo, hi, 0,1,2,3,4,5,6,7,8,9,10,11,12,13,14,15);
}
// B fragment (32x16, K x N), B[k][n] = src[n][k]: lane L owns column L&15,
// K {k0..k0+15} (L<16) or {k0+16..k0+31} (L>=16): one contiguous 32B load.
__device__ __forceinline__ v16bf load_b_frag(const __bf16* base, int col0,
                                             int stride, int k0, int lane) {
    int n = lane & 15, half = lane >> 4;
    const __bf16* rp = base + (size_t)(col0 + n) * stride + k0 + 16 * half;
    return *reinterpret_cast<const v16bf*>(rp);
}

// ---------------------------------------------------------------------------
// Kernels
// ---------------------------------------------------------------------------
__global__ void zero_out_kernel(float* out) {
    if (threadIdx.x == 0 && blockIdx.x == 0) out[0] = 0.0f;
}

// Gather sampled points into bf16, both point-major (Xh: [B][NP][c]) and
// channel-major (XhT: [B][c][NP]).  Padded rows (n >= N) are zero.
__global__ void gather_kernel(const float* __restrict__ tf,
                              const float* __restrict__ gf,
                              const int* __restrict__ idx,
                              int c, int hw, int N,
                              unsigned short* __restrict__ Xh,
                              unsigned short* __restrict__ Yh,
                              unsigned short* __restrict__ XhT,
                              unsigned short* __restrict__ YhT) {
    long t = (long)blockIdx.x * blockDim.x + threadIdx.x;
    long total = (long)BATCH * NP * c;
    if (t >= total) return;
    int ch = (int)(t % c);
    int n  = (int)((t / c) % NP);
    int b  = (int)(t / ((long)c * NP));
    float xv = 0.0f, yv = 0.0f;
    if (n < N) {
        int sp = idx ? idx[b * N + n] : n;
        size_t o = ((size_t)b * c + ch) * hw + sp;
        xv = tf[o];
        yv = gf[o];
    }
    unsigned short xb = f2bf(xv), yb = f2bf(yv);
    Xh [((size_t)b * NP + n) * c + ch] = xb;
    Yh [((size_t)b * NP + n) * c + ch] = yb;
    XhT[((size_t)b * c + ch) * NP + n] = xb;
    YhT[((size_t)b * c + ch) * NP + n] = yb;
}

// per-point L2 norms (from the bf16 data the GEMM actually sees)
__global__ void norm_kernel(const unsigned short* __restrict__ Xh,
                            const unsigned short* __restrict__ Yh,
                            int c, float* __restrict__ nx, float* __restrict__ ny) {
    int t = blockIdx.x * blockDim.x + threadIdx.x;
    if (t >= BATCH * NP) return;
    const unsigned short* xp = Xh + (size_t)t * c;
    const unsigned short* yp = Yh + (size_t)t * c;
    float sx = 0.0f, sy = 0.0f;
    for (int i = 0; i < c; ++i) {
        float a = bf2f(xp[i]); sx += a * a;
        float b = bf2f(yp[i]); sy += b * b;
    }
    nx[t] = sqrtf(sx);
    ny[t] = sqrtf(sy);
}

// per-channel means over true N (padded zeros contribute nothing)
__global__ void mu_kernel(const unsigned short* __restrict__ XhT,
                          const unsigned short* __restrict__ YhT,
                          int c, int N, float* __restrict__ mux, float* __restrict__ muy) {
    int t = blockIdx.x * blockDim.x + threadIdx.x;
    if (t >= BATCH * c) return;
    const unsigned short* xp = XhT + (size_t)t * NP;
    const unsigned short* yp = YhT + (size_t)t * NP;
    float sx = 0.0f, sy = 0.0f;
    for (int n = 0; n < NP; ++n) { sx += bf2f(xp[n]); sy += bf2f(yp[n]); }
    float inv = 1.0f / (float)N;
    mux[t] = sx * inv;
    muy[t] = sy * inv;
}

__global__ void init_kernel(unsigned* __restrict__ rowmin,
                            unsigned* __restrict__ colmin,
                            float* __restrict__ covsum) {
    int t = blockIdx.x * blockDim.x + threadIdx.x;
    if (t < BATCH * NP) { rowmin[t] = 0xFFFFFFFFu; colmin[t] = 0xFFFFFFFFu; }
    if (t < BATCH) covsum[t] = 0.0f;
}

// Each wave computes a 32x32 block of D = 1 - (X Y^T)/(|x|+eps)/(|y|+eps)
// as a 2x2 grid of WMMA tiles (A/B fragment reuse -> 4 WMMAs per 8 loads),
// then pre-reduces row/col mins before touching global atomics.
// Block = 128 threads = 4 waves, each owning a different 32-column block.
__global__ void __launch_bounds__(128)
sim_wmma_kernel(const unsigned short* __restrict__ Xh,
                const unsigned short* __restrict__ Yh,
                const float* __restrict__ nx, const float* __restrict__ ny,
                int c, int N,
                unsigned* __restrict__ rowmin, unsigned* __restrict__ colmin) {
    int lane = threadIdx.x & 31;
    int wv   = threadIdx.x >> 5;
    int b  = blockIdx.z;
    int i0 = blockIdx.x * 32;
    int j0 = (blockIdx.y * 4 + wv) * 32;
    const __bf16* X = reinterpret_cast<const __bf16*>(Xh) + (size_t)b * NP * c;
    const __bf16* Y = reinterpret_cast<const __bf16*>(Yh) + (size_t)b * NP * c;

    v8f acc[2][2] = {};
    for (int k0 = 0; k0 < c; k0 += 32) {
        v16bf a0 = load_a_frag(X, i0,      c, k0, lane);
        v16bf a1 = load_a_frag(X, i0 + 16, c, k0, lane);
        v16bf b0 = load_b_frag(Y, j0,      c, k0, lane);
        v16bf b1 = load_b_frag(Y, j0 + 16, c, k0, lane);
        acc[0][0] = WMMA_BF16(a0, b0, acc[0][0]);
        acc[0][1] = WMMA_BF16(a0, b1, acc[0][1]);
        acc[1][0] = WMMA_BF16(a1, b0, acc[1][0]);
        acc[1][1] = WMMA_BF16(a1, b1, acc[1][1]);
    }

    // D tile layout: VGPR r, lanes 0-15: M=r,N=lane; lanes 16-31: M=r+8,N=lane-16
    int n = lane & 15, hh = lane >> 4;
    float rowv[2][8];
#pragma unroll
    for (int a = 0; a < 2; ++a)
#pragma unroll
        for (int r = 0; r < 8; ++r) rowv[a][r] = BIGF;

#pragma unroll
    for (int b2 = 0; b2 < 2; ++b2) {
        int j = j0 + b2 * 16 + n;
        bool jv = (j < N);
        float dy = ny[b * NP + j] + EPSV;
        float cmin = BIGF;
#pragma unroll
        for (int a = 0; a < 2; ++a) {
#pragma unroll
            for (int r = 0; r < 8; ++r) {
                int i = i0 + a * 16 + r + 8 * hh;
                float dx = nx[b * NP + i] + EPSV;
                float d = 1.0f - acc[a][b2][r] / (dx * dy);
                if (jv) {
                    rowv[a][r] = fminf(rowv[a][r], d);
                    if (i < N) cmin = fminf(cmin, d);
                }
            }
        }
        if (jv && cmin < BIGF) atomicMin(&colmin[b * NP + j], f2key(cmin));
    }

    // row mins: reduce across the 16 lanes sharing the same half (same hh)
#pragma unroll
    for (int a = 0; a < 2; ++a) {
#pragma unroll
        for (int r = 0; r < 8; ++r) {
            float v = rowv[a][r];
            v = fminf(v, __shfl_xor(v, 1, 32));
            v = fminf(v, __shfl_xor(v, 2, 32));
            v = fminf(v, __shfl_xor(v, 4, 32));
            v = fminf(v, __shfl_xor(v, 8, 32));
            int i = i0 + a * 16 + r + 8 * hh;
            if (n == 0 && i < N && v < BIGF)
                atomicMin(&rowmin[b * NP + i], f2key(v));
        }
    }
}

// Each wave computes matching 32x32 blocks of X^T X and Y^T Y (2x2 WMMA tiles,
// channel-major copies -> contiguous fragments), converts to covariances with
// the mean correction, and accumulates sum |covX - covY|.
__global__ void __launch_bounds__(32)
gram_wmma_kernel(const unsigned short* __restrict__ XhT,
                 const unsigned short* __restrict__ YhT,
                 const float* __restrict__ mux, const float* __restrict__ muy,
                 int c, int N, float* __restrict__ covsum) {
    int lane = threadIdx.x;
    int b  = blockIdx.z;
    int i0 = blockIdx.x * 32;
    int j0 = blockIdx.y * 32;
    const __bf16* X = reinterpret_cast<const __bf16*>(XhT) + (size_t)b * c * NP;
    const __bf16* Y = reinterpret_cast<const __bf16*>(YhT) + (size_t)b * c * NP;

    v8f gx[2][2] = {}, gy[2][2] = {};
    for (int k0 = 0; k0 < NP; k0 += 32) {
        v16bf a0 = load_a_frag(X, i0,      NP, k0, lane);
        v16bf a1 = load_a_frag(X, i0 + 16, NP, k0, lane);
        v16bf b0 = load_b_frag(X, j0,      NP, k0, lane);
        v16bf b1 = load_b_frag(X, j0 + 16, NP, k0, lane);
        gx[0][0] = WMMA_BF16(a0, b0, gx[0][0]);
        gx[0][1] = WMMA_BF16(a0, b1, gx[0][1]);
        gx[1][0] = WMMA_BF16(a1, b0, gx[1][0]);
        gx[1][1] = WMMA_BF16(a1, b1, gx[1][1]);
        a0 = load_a_frag(Y, i0,      NP, k0, lane);
        a1 = load_a_frag(Y, i0 + 16, NP, k0, lane);
        b0 = load_b_frag(Y, j0,      NP, k0, lane);
        b1 = load_b_frag(Y, j0 + 16, NP, k0, lane);
        gy[0][0] = WMMA_BF16(a0, b0, gy[0][0]);
        gy[0][1] = WMMA_BF16(a0, b1, gy[0][1]);
        gy[1][0] = WMMA_BF16(a1, b0, gy[1][0]);
        gy[1][1] = WMMA_BF16(a1, b1, gy[1][1]);
    }

    float fN = (float)N, invNm1 = 1.0f / (float)(N - 1);
    int n = lane & 15, hh = lane >> 4;
    float local = 0.0f;
#pragma unroll
    for (int b2 = 0; b2 < 2; ++b2) {
        int j = j0 + b2 * 16 + n;
        float mxj = mux[b * c + j], myj = muy[b * c + j];
#pragma unroll
        for (int a = 0; a < 2; ++a) {
#pragma unroll
            for (int r = 0; r < 8; ++r) {
                int i = i0 + a * 16 + r + 8 * hh;
                float covx = (gx[a][b2][r] - fN * mux[b * c + i] * mxj) * invNm1;
                float covy = (gy[a][b2][r] - fN * muy[b * c + i] * myj) * invNm1;
                local += fabsf(covx - covy);
            }
        }
    }
    for (int off = 16; off > 0; off >>= 1)
        local += __shfl_xor(local, off, 32);
    if (lane == 0) atomicAdd(&covsum[b], local);
}

// Combine per batch: style = max(mean rowmin, mean colmin);
// moment = mean|mu_x-mu_y| + mean|covX-covY|;  out += (style+moment)/B
__global__ void final_kernel(const unsigned* __restrict__ rowmin,
                             const unsigned* __restrict__ colmin,
                             const float* __restrict__ mux,
                             const float* __restrict__ muy,
                             const float* __restrict__ covsum,
                             int c, int N, float* __restrict__ out) {
    int b = blockIdx.x, t = threadIdx.x;
    __shared__ float s1[256], s2[256], s3[256];
    float a1 = 0.0f, a2 = 0.0f, a3 = 0.0f;
    for (int i = t; i < N; i += 256) {
        a1 += key2f(rowmin[b * NP + i]);
        a2 += key2f(colmin[b * NP + i]);
    }
    for (int i = t; i < c; i += 256)
        a3 += fabsf(mux[b * c + i] - muy[b * c + i]);
    s1[t] = a1; s2[t] = a2; s3[t] = a3;
    __syncthreads();
    for (int s = 128; s > 0; s >>= 1) {
        if (t < s) { s1[t] += s1[t + s]; s2[t] += s2[t + s]; s3[t] += s3[t + s]; }
        __syncthreads();
    }
    if (t == 0) {
        float invN = 1.0f / (float)N;
        float style = fmaxf(s1[0] * invN, s2[0] * invN);
        float moment = s3[0] / (float)c + covsum[b] / ((float)c * (float)c);
        atomicAdd(out, (style + moment) * (1.0f / (float)BATCH));
    }
}

// ---------------------------------------------------------------------------
// Host launcher
// ---------------------------------------------------------------------------
extern "C" void kernel_launch(void* const* d_in, const int* in_sizes, int n_in,
                              void* d_out, int out_size, void* d_ws, size_t ws_size,
                              hipStream_t stream) {
    (void)in_sizes; (void)n_in; (void)out_size; (void)ws_size;
    struct Lv { int c, hw, N, tfi, gfi, idxi; };
    const Lv lv[4] = {
        { 64, 256 * 256, 1000, 0, 4,  8 },
        { 128, 128 * 128, 1000, 1, 5,  9 },
        { 256,  64 *  64, 1000, 2, 6, 10 },
        { 512,  32 *  32, 1024, 3, 7, -1 },
    };

    // workspace carve-out (reused across levels; stream-ordered launches)
    char* ws = (char*)d_ws;
    size_t off = 0;
    auto carve = [&](size_t bytes) -> void* {
        void* p = ws + off;
        off = (off + bytes + 255) & ~(size_t)255;
        return p;
    };
    const size_t maxC = 512;
    unsigned short* Xh  = (unsigned short*)carve((size_t)BATCH * NP * maxC * 2);
    unsigned short* Yh  = (unsigned short*)carve((size_t)BATCH * NP * maxC * 2);
    unsigned short* XhT = (unsigned short*)carve((size_t)BATCH * NP * maxC * 2);
    unsigned short* YhT = (unsigned short*)carve((size_t)BATCH * NP * maxC * 2);
    float*    nx     = (float*)carve((size_t)BATCH * NP * 4);
    float*    ny     = (float*)carve((size_t)BATCH * NP * 4);
    float*    mux    = (float*)carve((size_t)BATCH * maxC * 4);
    float*    muy    = (float*)carve((size_t)BATCH * maxC * 4);
    unsigned* rowmin = (unsigned*)carve((size_t)BATCH * NP * 4);
    unsigned* colmin = (unsigned*)carve((size_t)BATCH * NP * 4);
    float*    covsum = (float*)carve((size_t)BATCH * 4);

    float* out = (float*)d_out;
    zero_out_kernel<<<1, 32, 0, stream>>>(out);

    for (int l = 0; l < 4; ++l) {
        const float* tf = (const float*)d_in[lv[l].tfi];
        const float* gf = (const float*)d_in[lv[l].gfi];
        const int* idx = (lv[l].idxi >= 0) ? (const int*)d_in[lv[l].idxi] : nullptr;
        int c = lv[l].c, N = lv[l].N, hw = lv[l].hw;

        long total = (long)BATCH * NP * c;
        gather_kernel<<<(unsigned)((total + 255) / 256), 256, 0, stream>>>(
            tf, gf, idx, c, hw, N, Xh, Yh, XhT, YhT);
        norm_kernel<<<(BATCH * NP + 255) / 256, 256, 0, stream>>>(Xh, Yh, c, nx, ny);
        mu_kernel<<<(BATCH * c + 255) / 256, 256, 0, stream>>>(XhT, YhT, c, N, mux, muy);
        init_kernel<<<(BATCH * NP + 255) / 256, 256, 0, stream>>>(rowmin, colmin, covsum);

        // 32-row blocks x (4 waves of 32-column blocks) per block
        dim3 gsim(NP / 32, NP / 128, BATCH);
        sim_wmma_kernel<<<gsim, 128, 0, stream>>>(Xh, Yh, nx, ny, c, N, rowmin, colmin);

        dim3 ggram(c / 32, c / 32, BATCH);
        gram_wmma_kernel<<<ggram, 32, 0, stream>>>(XhT, YhT, mux, muy, c, N, covsum);

        final_kernel<<<BATCH, 256, 0, stream>>>(rowmin, colmin, mux, muy, covsum, c, N, out);
    }
}
